// FourierMMLayer_18090402251488
// MI455X (gfx1250) — matrix operand order repaired and checked by reading, not compiled, not run on hardware
//
#include <hip/hip_runtime.h>

#define BATCH 32
#define SEQ   512
#define HID   768
#define MROWS (BATCH * SEQ)   // 16384

typedef __attribute__((ext_vector_type(16))) _Float16 v16h;
typedef __attribute__((ext_vector_type(8)))  float    v8f;
typedef __attribute__((ext_vector_type(4)))  unsigned int v4u;
typedef __attribute__((ext_vector_type(8)))  int          v8i;
typedef __attribute__((ext_vector_type(4)))  int          v4i;

union Frag {
  v16h  v;
  uint4 q[2];
};

// ---------------- TDM helper ----------------
// Issue a 2D tensor_load_to_lds: tile (tileD1 rows x tileD0 cols) of f16,
// row stride `stride0` elements, into LDS at byte offset ldsOff with the
// row padding matching our [..][40] LDS tiles (64B data + 16B pad per row).
// D# packing per CDNA5 ISA section 8.3/8.4 (group0 128b, group1 256b; 2D ->
// groups 2/3 zero). data_size code 1 = 2 bytes. type = 2 ("image").
// This toolchain exposes the 6-arg builtin (extra zero-filled int32x8 group).
__device__ __forceinline__ void tdm_load_tile_f16(
    const _Float16* gsrc, unsigned ldsOff,
    unsigned tileD0, unsigned tileD1,
    unsigned long long stride0,
    unsigned tensorD0, unsigned tensorD1)
{
  unsigned long long ga = (unsigned long long)(const void*)gsrc;
  v4u g0;
  g0[0] = 1u;                                        // count=1 (valid user D#)
  g0[1] = ldsOff;                                    // lds_addr
  g0[2] = (unsigned)(ga & 0xffffffffu);              // global_addr[31:0]
  g0[3] = (unsigned)((ga >> 32) & 0x01ffffffu)       // global_addr[56:32]
        | (2u << 30);                                // type=2
  v8i g1;
  unsigned w0 = (1u << 16)                           // data_size = 2B
              | (1u << 20)                           // pad_enable
              | (3u << 22)                           // pad_interval: 16 DWORDs (64B)
              | (3u << 25);                          // pad_amount: 4 DWORDs (16B)
  g1[0] = (int)w0;
  g1[1] = (int)((tensorD0 & 0xffffu) << 16);         // atomic_barrier=0 | td0 lo
  g1[2] = (int)((tensorD0 >> 16) | ((tensorD1 & 0xffffu) << 16));
  g1[3] = (int)(((tensorD1 >> 16) & 0xffffu) | (tileD0 << 16));
  g1[4] = (int)(tileD1 & 0xffffu);                   // tile_dim1 | tile_dim2=0
  g1[5] = (int)(unsigned)(stride0 & 0xffffffffull);  // dim0 stride lo32
  g1[6] = (int)(unsigned)((stride0 >> 32) & 0xffffull); // stride hi16 | d1stride=0
  g1[7] = 0;
  v4i gz4 = {0, 0, 0, 0};
  v8i gz8 = {0, 0, 0, 0, 0, 0, 0, 0};
  __builtin_amdgcn_tensor_load_to_lds(g0, g1, gz4, gz4, gz8, 0);
}

__device__ __forceinline__ unsigned lds_offset(const void* p) {
  // LDS aperture: low 32 bits of the generic address are the LDS byte offset.
  return (unsigned)(unsigned long long)p;
}

// ---------------- conversion kernels ----------------

__global__ void cvt_f32_to_f16(const float* __restrict__ in,
                               _Float16* __restrict__ out, int n) {
  int i = blockIdx.x * blockDim.x + threadIdx.x;
  int stride = gridDim.x * blockDim.x;
  for (; i < n; i += stride) out[i] = (_Float16)in[i];
}

// complex128 buffer is interleaved (re, im) doubles.
__global__ void cvt_cplx_to_f16(const double* __restrict__ in,
                                _Float16* __restrict__ re,
                                _Float16* __restrict__ im,
                                float imSign, int n) {
  int i = blockIdx.x * blockDim.x + threadIdx.x;
  int stride = gridDim.x * blockDim.x;
  for (; i < n; i += stride) {
    re[i] = (_Float16)((float)in[2 * i]);
    im[i] = (_Float16)(imSign * (float)in[2 * i + 1]);
  }
}

// ---------------- stage 1: T1 = x @ Re(F_H), T2 = x @ Im(F_H) ----------------
// blockIdx.z selects {Re, Im}. M=16384, N=K=768.

__global__ __launch_bounds__(256) void fnet_stage1(
    const _Float16* __restrict__ X,
    const _Float16* __restrict__ RH,
    const _Float16* __restrict__ IH,
    _Float16* __restrict__ T1,
    _Float16* __restrict__ T2)
{
  const int K = HID, N = HID;
  const _Float16* __restrict__ Bm = (blockIdx.z == 0) ? RH : IH;
  _Float16* __restrict__ T        = (blockIdx.z == 0) ? T1 : T2;

  const int mBase = blockIdx.y * 128;
  const int nBase = blockIdx.x * 128;

  __shared__ _Float16 sA[128][40];   // A tile 128x32 (row-major, 16B row pad)
  __shared__ _Float16 sBt[128][40];  // B tile 32x128 stored transposed [n][k]

  const int tid  = threadIdx.x;
  const int lane = tid & 31;
  const int wave = tid >> 5;
  const int wm = wave & 3;   // 4 waves along M (32 rows each)
  const int wn = wave >> 2;  // 2 waves along N (64 cols each)

  const unsigned sAoff = lds_offset(&sA[0][0]);

  const v8f vzero = {0.f, 0.f, 0.f, 0.f, 0.f, 0.f, 0.f, 0.f};
  v8f acc[2][4];
#pragma unroll
  for (int i = 0; i < 2; ++i)
#pragma unroll
    for (int j = 0; j < 4; ++j) acc[i][j] = vzero;

  const int l15  = lane & 15;
  const int aRow = wm * 32 + l15;
  const int kb0  = (lane < 16) ? 0 : 8;   // ISA A-frag K base (lo half)
  const int bCol = wn * 64 + l15;
  const int kbb  = (lane < 16) ? 0 : 8;   // ISA B-frag K base

  for (int k0 = 0; k0 < K; k0 += 32) {
    // ---- A tile via Tensor Data Mover (async, TENSORcnt) ----
    if (wave == 0)
      tdm_load_tile_f16(X + (size_t)mBase * K + k0, sAoff,
                        /*tile cols*/32, /*tile rows*/128,
                        /*row stride*/(unsigned long long)K,
                        /*tensor d0*/K, /*tensor d1*/MROWS);

    // ---- B tile: global -> LDS transposed (all threads) ----
#pragma unroll
    for (int it = 0; it < 2; ++it) {
      int idx = tid + it * 256;
      int r = idx >> 4, s = idx & 15;
      union { uint4 q; _Float16 h[8]; } u;
      u.q = *(const uint4*)(Bm + (size_t)(k0 + r) * N + (nBase + s * 8));
#pragma unroll
      for (int j = 0; j < 8; ++j) sBt[s * 8 + j][r] = u.h[j];
    }
    if (k0 + 32 < K)  // warm next B tile (global_prefetch_b8)
      __builtin_prefetch(Bm + (size_t)(k0 + 32 + (tid >> 4)) * N + nBase, 0, 1);

    if (wave == 0) __builtin_amdgcn_s_wait_tensorcnt(0);
    __syncthreads();

    Frag a[2], b[4];
#pragma unroll
    for (int i = 0; i < 2; ++i) {
      a[i].q[0] = *(const uint4*)&sA[aRow + i * 16][kb0];       // K kb0..kb0+7
      a[i].q[1] = *(const uint4*)&sA[aRow + i * 16][kb0 + 16];  // K kb0+16..+23
    }
#pragma unroll
    for (int j = 0; j < 4; ++j) {
      b[j].q[0] = *(const uint4*)&sBt[bCol + j * 16][kbb];      // K kbb..kbb+7
      b[j].q[1] = *(const uint4*)&sBt[bCol + j * 16][kbb + 16]; // K kbb+16..+23
    }
#pragma unroll
    for (int i = 0; i < 2; ++i)
#pragma unroll
      for (int j = 0; j < 4; ++j)
        acc[i][j] = __builtin_amdgcn_wmma_f32_16x16x32_f16(
            false, a[i].v, false, b[j].v, (short)0, acc[i][j], false, false);
    __syncthreads();  // orders next TDM write vs all waves' fragment reads
  }

  // store accumulators as f16 (C/D layout: VGPR r -> row r (+8 for lanes 16-31))
  const int mOff = mBase + wm * 32 + ((lane < 16) ? 0 : 8);
#pragma unroll
  for (int i = 0; i < 2; ++i)
#pragma unroll
    for (int j = 0; j < 4; ++j) {
      const int n = nBase + wn * 64 + j * 16 + l15;
#pragma unroll
      for (int r = 0; r < 8; ++r)
        T[(size_t)(mOff + i * 16 + r) * N + n] = (_Float16)acc[i][j][r];
    }
}

// ---------------- stage 2: out_b = Re(F_S)@T1_b + (-Im(F_S))@T2_b ----------------
// Per batch: M=512, K=512, N=768. f32 output.

__global__ __launch_bounds__(256) void fnet_stage2(
    const _Float16* __restrict__ RS,
    const _Float16* __restrict__ ISn,
    const _Float16* __restrict__ T1,
    const _Float16* __restrict__ T2,
    float* __restrict__ out)
{
  const int K = SEQ, N = HID;
  const int b = blockIdx.z;
  const _Float16* __restrict__ t1 = T1 + (size_t)b * SEQ * HID;
  const _Float16* __restrict__ t2 = T2 + (size_t)b * SEQ * HID;
  float* __restrict__ ob = out + (size_t)b * SEQ * HID;

  const int mBase = blockIdx.y * 128;
  const int nBase = blockIdx.x * 128;

  __shared__ _Float16 sA1[128][40];
  __shared__ _Float16 sA2[128][40];
  __shared__ _Float16 sB1[128][40];  // transposed [n][k]
  __shared__ _Float16 sB2[128][40];

  const int tid  = threadIdx.x;
  const int lane = tid & 31;
  const int wave = tid >> 5;
  const int wm = wave & 3;
  const int wn = wave >> 2;

  const unsigned sA1off = lds_offset(&sA1[0][0]);
  const unsigned sA2off = lds_offset(&sA2[0][0]);

  const v8f vzero = {0.f, 0.f, 0.f, 0.f, 0.f, 0.f, 0.f, 0.f};
  v8f acc[2][4];
#pragma unroll
  for (int i = 0; i < 2; ++i)
#pragma unroll
    for (int j = 0; j < 4; ++j) acc[i][j] = vzero;

  const int l15  = lane & 15;
  const int aRow = wm * 32 + l15;
  const int kb0  = (lane < 16) ? 0 : 8;
  const int bCol = wn * 64 + l15;
  const int kbb  = (lane < 16) ? 0 : 8;

  for (int k0 = 0; k0 < K; k0 += 32) {
    // ---- two A tiles via TDM, issued by two different waves (parallel DMA) ----
    if (wave == 0)
      tdm_load_tile_f16(RS + (size_t)mBase * K + k0, sA1off,
                        32, 128, (unsigned long long)K, K, SEQ);
    if (wave == 1)
      tdm_load_tile_f16(ISn + (size_t)mBase * K + k0, sA2off,
                        32, 128, (unsigned long long)K, K, SEQ);

    // ---- B tiles: global -> LDS transposed ----
#pragma unroll
    for (int it = 0; it < 2; ++it) {
      int idx = tid + it * 256;
      int r = idx >> 4, s = idx & 15;
      union { uint4 q; _Float16 h[8]; } u1, u2;
      u1.q = *(const uint4*)(t1 + (size_t)(k0 + r) * N + (nBase + s * 8));
      u2.q = *(const uint4*)(t2 + (size_t)(k0 + r) * N + (nBase + s * 8));
#pragma unroll
      for (int j = 0; j < 8; ++j) {
        sB1[s * 8 + j][r] = u1.h[j];
        sB2[s * 8 + j][r] = u2.h[j];
      }
    }
    if (k0 + 32 < K) {
      __builtin_prefetch(t1 + (size_t)(k0 + 32 + (tid >> 4)) * N + nBase, 0, 1);
      __builtin_prefetch(t2 + (size_t)(k0 + 32 + (tid >> 4)) * N + nBase, 0, 1);
    }

    if (wave < 2) __builtin_amdgcn_s_wait_tensorcnt(0);
    __syncthreads();

    Frag a1[2], a2[2];
#pragma unroll
    for (int i = 0; i < 2; ++i) {
      a1[i].q[0] = *(const uint4*)&sA1[aRow + i * 16][kb0];
      a1[i].q[1] = *(const uint4*)&sA1[aRow + i * 16][kb0 + 16];
      a2[i].q[0] = *(const uint4*)&sA2[aRow + i * 16][kb0];
      a2[i].q[1] = *(const uint4*)&sA2[aRow + i * 16][kb0 + 16];
    }
#pragma unroll
    for (int j = 0; j < 4; ++j) {
      Frag bf1, bf2;
      bf1.q[0] = *(const uint4*)&sB1[bCol + j * 16][kbb];
      bf1.q[1] = *(const uint4*)&sB1[bCol + j * 16][kbb + 16];
      bf2.q[0] = *(const uint4*)&sB2[bCol + j * 16][kbb];
      bf2.q[1] = *(const uint4*)&sB2[bCol + j * 16][kbb + 16];
#pragma unroll
      for (int i = 0; i < 2; ++i) {
        acc[i][j] = __builtin_amdgcn_wmma_f32_16x16x32_f16(
            false, a1[i].v, false, bf1.v, (short)0, acc[i][j], false, false);
        acc[i][j] = __builtin_amdgcn_wmma_f32_16x16x32_f16(
            false, a2[i].v, false, bf2.v, (short)0, acc[i][j], false, false);
      }
    }
    __syncthreads();
  }

  const int mOff = mBase + wm * 32 + ((lane < 16) ? 0 : 8);
#pragma unroll
  for (int i = 0; i < 2; ++i)
#pragma unroll
    for (int j = 0; j < 4; ++j) {
      const int n = nBase + wn * 64 + j * 16 + l15;
#pragma unroll
      for (int r = 0; r < 8; ++r)
        ob[(size_t)(mOff + i * 16 + r) * N + n] = acc[i][j][r];
    }
}

// ---------------- launch ----------------

extern "C" void kernel_launch(void* const* d_in, const int* in_sizes, int n_in,
                              void* d_out, int out_size, void* d_ws, size_t ws_size,
                              hipStream_t stream) {
  (void)in_sizes; (void)n_in; (void)out_size; (void)ws_size;

  const float*  x     = (const float*)d_in[0];
  const double* dft_s = (const double*)d_in[1];  // [512,512] complex128 interleaved
  const double* dft_h = (const double*)d_in[2];  // [768,768] complex128 interleaved

  // workspace layout (all offsets 256B aligned)
  char* ws = (char*)d_ws;
  _Float16* xh  = (_Float16*)(ws + 0);          // 16384*768 f16 = 25,165,824 B
  _Float16* RH  = (_Float16*)(ws + 25165824);   // 768*768 f16
  _Float16* IH  = (_Float16*)(ws + 26345472);   // 768*768 f16
  _Float16* RS  = (_Float16*)(ws + 27525120);   // 512*512 f16
  _Float16* ISn = (_Float16*)(ws + 28049408);   // 512*512 f16 (negated Im)
  _Float16* T1  = (_Float16*)(ws + 28573696);   // 16384*768 f16
  _Float16* T2  = (_Float16*)(ws + 53739520);   // 16384*768 f16

  cvt_f32_to_f16<<<1024, 256, 0, stream>>>(x, xh, MROWS * HID);
  cvt_cplx_to_f16<<<576, 256, 0, stream>>>(dft_h, RH, IH, 1.0f, HID * HID);
  cvt_cplx_to_f16<<<256, 256, 0, stream>>>(dft_s, RS, ISn, -1.0f, SEQ * SEQ);

  fnet_stage1<<<dim3(HID / 128, MROWS / 128, 2), 256, 0, stream>>>(
      xh, RH, IH, T1, T2);
  fnet_stage2<<<dim3(HID / 128, SEQ / 128, BATCH), 256, 0, stream>>>(
      RS, ISn, T1, T2, (float*)d_out);
}
